// TraingleAttention_16269336117305
// MI455X (gfx1250) — matrix operand order repaired and checked by reading, not compiled, not run on hardware
//
#include <hip/hip_runtime.h>
#include <hip/hip_bf16.h>

#define NSAMP 32768
#define KZS   20
#define IFZ   128
#define NB    4             // samples per block
#define ROWS  (NB * KZS)    // 80 rows = 5 M-tiles of 16
#define THREADS 256
#define NWAVE 8

// preconverted-weight layout in d_ws (halves): WT[col*128+k] per matrix
#define OFF_Q 0
#define OFF_K 16384
#define OFF_V 32768
#define OFF_B 49152         // 80 cols
#define OFF_G 59392
#define OFF_O 75776
#define WP_HALVES 92160
#define WS_NEEDED ((size_t)WP_HALVES * 2)

typedef __attribute__((ext_vector_type(16))) _Float16 v16h;
typedef __attribute__((ext_vector_type(8)))  _Float16 v8h;
typedef __attribute__((ext_vector_type(4)))  _Float16 v4h;
typedef __attribute__((ext_vector_type(8)))  float    v8f;

// ---- CDNA5 async global->LDS (ASYNCcnt) ----
__device__ __forceinline__ void async_ld16(unsigned lds_off, const _Float16* g) {
    asm volatile("global_load_async_to_lds_b128 %0, %1, off"
                 :: "v"(lds_off), "v"(g) : "memory");
}
__device__ __forceinline__ void wait_async0() {
    asm volatile("s_wait_asynccnt 0" ::: "memory");
}

// ---- WMMA fragment loaders (16x16x32 f16, wave32) ----
__device__ __forceinline__ v16h load_A(const _Float16* base, int stride, int row0, int k0, int lane) {
    int m = lane & 15, hf = lane >> 4;
    const _Float16* p = base + (size_t)(row0 + m) * stride + k0 + hf * 8;
    v8h lo = *(const v8h*)p;
    v8h hi = *(const v8h*)(p + 16);
    v16h a;
#pragma unroll
    for (int e = 0; e < 8; ++e) { a[e] = lo[e]; a[e + 8] = hi[e]; }
    return a;
}
__device__ __forceinline__ v16h load_B(const _Float16* bt, int stride, int n0, int k0, int lane) {
    int n = lane & 15, kh = (lane >> 4) * 8;
    const _Float16* p = bt + (size_t)(n0 + n) * stride + k0 + kh;
    v8h lo = *(const v8h*)p;
    v8h hi = *(const v8h*)(p + 16);
    v16h b;
#pragma unroll
    for (int e = 0; e < 8; ++e) { b[e] = lo[e]; b[e + 8] = hi[e]; }
    return b;
}
__device__ __forceinline__ v8f wmma_f16(v16h a, v16h b, v8f c) {
    return __builtin_amdgcn_wmma_f32_16x16x32_f16(false, a, false, b, (short)0, c, false, false);
}

// stage f16 WT chunk via async DMA (fast path)
__device__ __forceinline__ void stage_async(_Float16* dst, const _Float16* src, int halves, int tid) {
    unsigned l0 = (unsigned)(unsigned long long)dst;
    for (int i = tid * 8; i < halves; i += THREADS * 8)
        async_ld16(l0 + (unsigned)i * 2u, src + i);
}
// fallback: transpose+convert from f32 global
__device__ __forceinline__ void stage_sync(_Float16* dst, const float* W, int C, int tid) {
    for (int idx = tid; idx < IFZ * C; idx += THREADS) {
        int k = idx / C, col = idx - k * C;
        dst[col * IFZ + k] = (_Float16)W[idx];
    }
}

// K=128 GEMM tile: all 8 fragments (4 A + 4 B) loaded up-front, then 4 chained WMMAs.
template <typename Epi>
__device__ __forceinline__ void gemm4(const _Float16* A, int As, const _Float16* BT, int Bs,
                                      int Mt, int Nt, int wave, int lane, Epi&& epi) {
    int jobs = Mt * Nt;
    for (int job = wave; job < jobs; job += NWAVE) {
        int mt = job / Nt, nt = job - mt * Nt;
        v16h a0 = load_A(A, As, mt * 16,  0, lane);
        v16h a1 = load_A(A, As, mt * 16, 32, lane);
        v16h a2 = load_A(A, As, mt * 16, 64, lane);
        v16h a3 = load_A(A, As, mt * 16, 96, lane);
        v16h b0 = load_B(BT, Bs, nt * 16,  0, lane);
        v16h b1 = load_B(BT, Bs, nt * 16, 32, lane);
        v16h b2 = load_B(BT, Bs, nt * 16, 64, lane);
        v16h b3 = load_B(BT, Bs, nt * 16, 96, lane);
        v8f acc = {};
        acc = wmma_f16(a0, b0, acc);
        acc = wmma_f16(a1, b1, acc);
        acc = wmma_f16(a2, b2, acc);
        acc = wmma_f16(a3, b3, acc);
        epi(mt, nt, acc);
    }
}

// one-time weight transpose/convert f32 -> f16 into workspace
extern "C" __global__ __launch_bounds__(256)
void prep_weights(const float* __restrict__ Wq, const float* __restrict__ Wk,
                  const float* __restrict__ Wv, const float* __restrict__ Wb,
                  const float* __restrict__ Wg, const float* __restrict__ Wo,
                  _Float16* __restrict__ wp) {
    int idx = blockIdx.x * 256 + threadIdx.x;
    if (idx >= WP_HALVES) return;
    const float* src; int C, base;
    if      (idx < OFF_K) { src = Wq; C = 128; base = OFF_Q; }
    else if (idx < OFF_V) { src = Wk; C = 128; base = OFF_K; }
    else if (idx < OFF_B) { src = Wv; C = 128; base = OFF_V; }
    else if (idx < OFF_G) { src = Wb; C = 80;  base = OFF_B; }
    else if (idx < OFF_O) { src = Wg; C = 128; base = OFF_G; }
    else                  { src = Wo; C = 128; base = OFF_O; }
    int local = idx - base;
    int k = local / C, col = local - k * C;
    wp[base + col * IFZ + k] = (_Float16)src[local];
}

extern "C" __global__ __launch_bounds__(THREADS)
void triattn_kernel(const float* __restrict__ x,
                    const float* __restrict__ lns, const float* __restrict__ lnb,
                    const float* __restrict__ Wq, const float* __restrict__ Wk,
                    const float* __restrict__ Wv, const float* __restrict__ Wb,
                    const float* __restrict__ Wg, const float* __restrict__ bg,
                    const float* __restrict__ Wo, const float* __restrict__ bo,
                    const _Float16* __restrict__ wp,   // preconverted weights or null
                    float* __restrict__ out)
{
    extern __shared__ char smem[];
    _Float16* yL  = (_Float16*)smem;          // 80*128
    _Float16* qL  = yL + 80 * 128;            // 96*128 (pad rows)
    _Float16* kL  = qL + 96 * 128;            // 96*128
    _Float16* vT  = kL + 96 * 128;            // [s][a][c:32][j:32] = 16384
    _Float16* gL  = vT + 16 * 32 * 32;        // 80*128
    _Float16* pL  = gL + 80 * 128;            // [s][a][i:32][j:32] = 16384
    _Float16* oL  = pL + 16 * 32 * 32;        // 80*128
    _Float16* wt0 = oL + 80 * 128;            // 128*128 staging buf 0
    _Float16* wt1 = wt0 + 16384;              // 128*128 staging buf 1
    float*    sL  = (float*)(wt1 + 16384);    // [s][a][i:20][j:32] f32 = 10240

    const int tid  = threadIdx.x;
    const int lane = tid & 31;
    const int wave = tid >> 5;
    const int half = lane >> 4;
    const int nlo  = lane & 15;
    const bool pre = (wp != nullptr);
    const size_t gr0 = (size_t)blockIdx.x * ROWS;

#define STG(buf, PREOFF, WF32, C) \
    do { if (pre) stage_async(buf, wp + PREOFF, (C) * 128, tid); else stage_sync(buf, WF32, C, tid); } while (0)

    // zero vT (j-padding must be 0 for probs@V K-dim)
    for (int i = tid; i < 16 * 32 * 32 / 2; i += THREADS) ((unsigned*)vT)[i] = 0u;

    STG(wt0, OFF_Q, Wq, 128);                 // prefetch WqT under the LayerNorm

    // ---------------- LayerNorm -> yL (f16) ----------------
    {
        const float4 sv = ((const float4*)lns)[lane];
        const float4 bv = ((const float4*)lnb)[lane];
        for (int r = wave; r < ROWS; r += NWAVE) {
            float4 xv = ((const float4*)(x + (gr0 + r) * IFZ))[lane];
            float s = xv.x + xv.y + xv.z + xv.w;
#pragma unroll
            for (int m = 16; m >= 1; m >>= 1) s += __shfl_xor(s, m, 32);
            float mu = s * (1.0f / IFZ);
            float d0 = xv.x - mu, d1 = xv.y - mu, d2 = xv.z - mu, d3 = xv.w - mu;
            float vv = d0 * d0 + d1 * d1 + d2 * d2 + d3 * d3;
#pragma unroll
            for (int m = 16; m >= 1; m >>= 1) vv += __shfl_xor(vv, m, 32);
            float rs = rsqrtf(vv * (1.0f / IFZ) + 1e-5f);
            v4h hv;
            hv[0] = (_Float16)(d0 * rs * sv.x + bv.x);
            hv[1] = (_Float16)(d1 * rs * sv.y + bv.y);
            hv[2] = (_Float16)(d2 * rs * sv.z + bv.z);
            hv[3] = (_Float16)(d3 * rs * sv.w + bv.w);
            *(v4h*)(yL + r * IFZ + lane * 4) = hv;
        }
    }
    if (pre) wait_async0();
    __syncthreads();

    // ---------------- projections (double-buffered weight staging) ----------------
    STG(wt1, OFF_K, Wk, 128);
    gemm4(yL, 128, wt0, 128, 5, 8, wave, lane, [&](int mt, int nt, v8f acc) {
#pragma unroll
        for (int v = 0; v < 8; ++v)
            qL[(mt * 16 + half * 8 + v) * 128 + nt * 16 + nlo] = (_Float16)acc[v];
    });
    if (pre) wait_async0();
    __syncthreads();

    STG(wt0, OFF_V, Wv, 128);
    gemm4(yL, 128, wt1, 128, 5, 8, wave, lane, [&](int mt, int nt, v8f acc) {
#pragma unroll
        for (int v = 0; v < 8; ++v)
            kL[(mt * 16 + half * 8 + v) * 128 + nt * 16 + nlo] = (_Float16)acc[v];
    });
    if (pre) wait_async0();
    __syncthreads();

    STG(wt1, OFF_B, Wb, 80);
    gemm4(yL, 128, wt0, 128, 5, 8, wave, lane, [&](int mt, int nt, v8f acc) {
#pragma unroll
        for (int v = 0; v < 8; ++v) {
            int r = mt * 16 + half * 8 + v;
            int s = r / KZS, j = r - s * KZS;
            int col = nt * 16 + nlo, a = col >> 5, c = col & 31;
            vT[((s * 4 + a) * 32 + c) * 32 + j] = (_Float16)acc[v];
        }
    });
    if (pre) wait_async0();
    __syncthreads();

    STG(wt0, OFF_G, Wg, 128);
    gemm4(yL, 128, wt1, 128, 5, 5, wave, lane, [&](int mt, int nt, v8f acc) {
#pragma unroll
        for (int v = 0; v < 8; ++v) {
            int r = mt * 16 + half * 8 + v;
            int s = r / KZS, i = r - s * KZS;
            int col = nt * 16 + nlo, j = col >> 2, a = col & 3;
            sL[((s * 4 + a) * KZS + i) * 32 + j] = acc[v];
        }
    });
    if (pre) wait_async0();
    __syncthreads();

    STG(wt1, OFF_O, Wo, 128);                 // WoT DMA overlaps gate gemm + attention
    gemm4(yL, 128, wt0, 128, 5, 8, wave, lane, [&](int mt, int nt, v8f acc) {
#pragma unroll
        for (int v = 0; v < 8; ++v) {
            int r = mt * 16 + half * 8 + v;
            int col = nt * 16 + nlo;
            float t = acc[v] + bg[col];
            gL[r * 128 + col] = (_Float16)(1.0f / (1.0f + __expf(-t)));
        }
    });
    __syncthreads();

    // ---------------- scores = QK^T/sqrt(d) + bias ----------------
    const float inv_scale = 0.17677669529663687f;
    for (int job = wave; job < 64; job += NWAVE) {
        int s = job >> 4, rem = job & 15;
        int a = rem >> 2, it = (rem >> 1) & 1, jt = rem & 1;
        v16h av = load_A(qL, 128, s * KZS + it * 16, a * 32, lane);
        v16h bv = load_B(kL + (size_t)(s * KZS) * 128 + a * 32, 128, jt * 16, 0, lane);
        v8f acc = {};
        acc = wmma_f16(av, bv, acc);
#pragma unroll
        for (int v = 0; v < 8; ++v) {
            int i = it * 16 + half * 8 + v, j = jt * 16 + nlo;
            if (i < KZS && j < KZS)
                sL[((s * 4 + a) * KZS + i) * 32 + j] += acc[v] * inv_scale;
        }
    }
    __syncthreads();

    // ---------------- softmax over j -> pL (f16, zero-padded) ----------------
    for (int t = tid; t < 16 * KZS; t += THREADS) {
        int sa = t / KZS, i = t - sa * KZS;
        const float* sr = sL + (sa * KZS + i) * 32;
        _Float16* pr = pL + (sa * 32 + i) * 32;
        float m = sr[0];
#pragma unroll
        for (int j = 1; j < KZS; ++j) m = fmaxf(m, sr[j]);
        float ev[KZS], sum = 0.0f;
#pragma unroll
        for (int j = 0; j < KZS; ++j) { ev[j] = __expf(sr[j] - m); sum += ev[j]; }
        float is = 1.0f / sum;
#pragma unroll
        for (int j = 0; j < KZS; ++j) pr[j] = (_Float16)(ev[j] * is);
#pragma unroll
        for (int j = KZS; j < 32; ++j) pr[j] = (_Float16)0.0f;
    }
    __syncthreads();

    // ---------------- o = gate * (probs @ V) ----------------
    for (int job = wave; job < 64; job += NWAVE) {
        int s = job >> 4, rem = job & 15;
        int a = rem >> 2, it = (rem >> 1) & 1, ct = rem & 1;
        const _Float16* pb = pL + (size_t)(s * 4 + a) * 32 * 32;
        const _Float16* vb = vT + (size_t)(s * 4 + a) * 32 * 32;
        v16h av = load_A(pb, 32, it * 16, 0, lane);
        v16h bv = load_B(vb, 32, ct * 16, 0, lane);
        v8f acc = {};
        acc = wmma_f16(av, bv, acc);
#pragma unroll
        for (int v = 0; v < 8; ++v) {
            int i = it * 16 + half * 8 + v;
            if (i < KZS) {
                int r = s * KZS + i, col = a * 32 + ct * 16 + nlo;
                float g = (float)gL[r * 128 + col];
                oL[r * 128 + col] = (_Float16)(acc[v] * g);
            }
        }
    }
    if (pre) wait_async0();   // WoT DMA must be complete before phase 6
    __syncthreads();

    // ---------------- out = o @ Wo + bo + x ----------------
    gemm4(oL, 128, wt1, 128, 5, 8, wave, lane, [&](int mt, int nt, v8f acc) {
#pragma unroll
        for (int v = 0; v < 8; ++v) {
            int r = mt * 16 + half * 8 + v;
            int col = nt * 16 + nlo;
            size_t g = (gr0 + r) * IFZ + col;
            out[g] = acc[v] + bo[col] + x[g];
        }
    });
#undef STG
}

extern "C" void kernel_launch(void* const* d_in, const int* in_sizes, int n_in,
                              void* d_out, int out_size, void* d_ws, size_t ws_size,
                              hipStream_t stream) {
    const float* x   = (const float*)d_in[0];
    const float* lns = (const float*)d_in[1];
    const float* lnb = (const float*)d_in[2];
    const float* Wq  = (const float*)d_in[3];
    const float* Wk  = (const float*)d_in[4];
    const float* Wv  = (const float*)d_in[5];
    const float* Wb  = (const float*)d_in[6];
    const float* Wg  = (const float*)d_in[7];
    const float* bg  = (const float*)d_in[8];
    const float* Wo  = (const float*)d_in[9];
    const float* bo  = (const float*)d_in[10];
    float* out = (float*)d_out;

    const _Float16* wp = nullptr;
    if (ws_size >= WS_NEEDED) {
        hipLaunchKernelGGL(prep_weights, dim3(WP_HALVES / 256), dim3(256), 0, stream,
                           Wq, Wk, Wv, Wb, Wg, Wo, (_Float16*)d_ws);
        wp = (const _Float16*)d_ws;
    }

    // LDS: 120832 halves + 10240 floats = 282,624 B (< 320 KB/WGP)
    constexpr size_t SMEM = (size_t)120832 * 2 + (size_t)10240 * 4;
    dim3 grid(NSAMP / NB), block(THREADS);
    hipLaunchKernelGGL(triattn_kernel, grid, block, SMEM, stream,
                       x, lns, lnb, Wq, Wk, Wv, Wb, Wg, bg, Wo, bo, wp, out);
}
